// AttentionBlock_87385404604750
// MI455X (gfx1250) — matrix operand order repaired and checked by reading, not compiled, not run on hardware
//
#include <hip/hip_runtime.h>
#include <hip/hip_bf16.h>

// ---------------------------------------------------------------------------
// Problem constants (reference: B=4, H=W=64, F=256, FQ=32)
// ---------------------------------------------------------------------------
#define F_DIM   256
#define FQ_DIM  32
#define B_DIM   4
#define N_TOK   4096           // 64*64
#define BN_TOK  16384          // B*N

// LDS row strides (elements) padded to avoid bank conflicts on b128 reads.
#define KT_STR  40
#define VT_STR  40
#define PT_STR  40

typedef __attribute__((ext_vector_type(16))) __bf16 v16bf;
typedef __attribute__((ext_vector_type(8)))  __bf16 bf16x8;
typedef __attribute__((ext_vector_type(8)))  float  v8f;

// native f32 -> bf16 conversion (v_cvt_* on gfx1250)
__device__ __forceinline__ __bf16 f2bf(float f) { return (__bf16)f; }

__device__ __forceinline__ bf16x8 load8(const __bf16* p) {
    return *(const bf16x8*)p;   // 16B-aligned contiguous -> b128 load
}

// ===========================================================================
// Prep kernel A: x (f32) -> xb (bf16), vectorized.
// grid: BN*F/(256*4) blocks of 256 threads.
// ===========================================================================
__global__ __launch_bounds__(256)
void cvt_x_kernel(const float* __restrict__ x, unsigned short* __restrict__ xb_u)
{
    __bf16* xb = (__bf16*)xb_u;
    const size_t i = ((size_t)blockIdx.x * 256 + threadIdx.x) * 4;
    const float4 v = *(const float4*)(x + i);
    xb[i + 0] = f2bf(v.x);
    xb[i + 1] = f2bf(v.y);
    xb[i + 2] = f2bf(v.z);
    xb[i + 3] = f2bf(v.w);
}

// ===========================================================================
// Prep kernel B: transpose + convert weights to bf16.
//   wt layout: WqT[32][256] | WkT[32][256] | WvT[256][256] | WoT[256][256]
// grid: (256 cols, 4 matrices) blocks of 256 threads (thread = k index).
// ===========================================================================
__global__ __launch_bounds__(256)
void wt_kernel(const float* __restrict__ Wq, const float* __restrict__ Wk,
               const float* __restrict__ Wv, const float* __restrict__ Wo,
               unsigned short* __restrict__ wt_u)
{
    __bf16* wt = (__bf16*)wt_u;
    const int m   = blockIdx.y;
    const int col = blockIdx.x;
    const int k   = threadIdx.x;

    const float* W; int D; size_t off;
    if (m == 0)      { W = Wq; D = FQ_DIM; off = 0; }
    else if (m == 1) { W = Wk; D = FQ_DIM; off = (size_t)32 * 256; }
    else if (m == 2) { W = Wv; D = F_DIM;  off = (size_t)64 * 256; }
    else             { W = Wo; D = F_DIM;  off = (size_t)(64 + 256) * 256; }

    if (col < D)
        wt[off + (size_t)col * 256 + k] = f2bf(W[(size_t)k * D + col]);
}

// ===========================================================================
// Kernel 1: fused QKV projection (all-bf16 operands, b128 loads only).
// Grid: (BN/16, 20) blocks of 32 threads (1 wave -> one 16x16 output tile).
// chunks: 0-1 -> q, 2-3 -> k, 4-19 -> v
// ===========================================================================
__global__ __launch_bounds__(32)
void qkv_kernel(const unsigned short* __restrict__ xb_u,
                const unsigned short* __restrict__ wt_u,
                const float* __restrict__ bq, const float* __restrict__ bk,
                const float* __restrict__ bv,
                unsigned short* __restrict__ q_u,
                unsigned short* __restrict__ k_u,
                unsigned short* __restrict__ v_u)
{
    const __bf16* xb = (const __bf16*)xb_u;
    const __bf16* wt = (const __bf16*)wt_u;

    const int rt   = blockIdx.x;          // 16-row tile
    const int c    = blockIdx.y;          // output chunk 0..19
    const int lane = threadIdx.x;
    const int half = lane >> 4;           // 0/1
    const int lrow = lane & 15;           // A row / C column

    const __bf16* WT; const float* bias; __bf16* outp; int D; int colbase;
    if (c < 2)       { WT = wt;                     bias = bq; outp = (__bf16*)q_u; D = FQ_DIM; colbase = c * 16; }
    else if (c < 4)  { WT = wt + (size_t)32 * 256;  bias = bk; outp = (__bf16*)k_u; D = FQ_DIM; colbase = (c - 2) * 16; }
    else             { WT = wt + (size_t)64 * 256;  bias = bv; outp = (__bf16*)v_u; D = F_DIM;  colbase = (c - 4) * 16; }

    const int row0 = rt * 16;
    const __bf16* ap0 = xb + (size_t)(row0 + lrow) * F_DIM + half * 8;
    const __bf16* bp0 = WT + (size_t)(colbase + lrow) * 256 + half * 8;

    v8f acc = {};
    #pragma unroll
    for (int k0 = 0; k0 < F_DIM; k0 += 32) {
        v16bf a, b;
        bf16x8 alo = load8(ap0 + k0);
        bf16x8 ahi = load8(ap0 + k0 + 16);
        bf16x8 blo = load8(bp0 + k0);
        bf16x8 bhi = load8(bp0 + k0 + 16);
        #pragma unroll
        for (int i = 0; i < 8; ++i) {
            a[i] = alo[i]; a[8 + i] = ahi[i];
            b[i] = blo[i]; b[8 + i] = bhi[i];
        }
        acc = __builtin_amdgcn_wmma_f32_16x16x32_bf16(false, a, false, b,
                                                      (short)0, acc, false, false);
    }

    const float bb = bias[colbase + lrow];
    #pragma unroll
    for (int i = 0; i < 8; ++i) {
        float o = acc[i] + bb;
        o = o > 0.f ? o : 0.f;
        const int m = i + half * 8;
        outp[(size_t)(row0 + m) * D + colbase + lrow] = f2bf(o);
    }
}

// ===========================================================================
// Kernel 2: flash attention (unscaled softmax, as reference).
// Grid: (N/16, B) blocks of 512 threads (16 waves).
//   wave 0: S = Q K^T (2 WMMAs), online softmax, P+alpha -> LDS
//   all 16 waves: rescale accumulators, O += P V (1 WMMA each, 16ch slice)
// ===========================================================================
__global__ __launch_bounds__(512)
void attn_kernel(const unsigned short* __restrict__ q_u,
                 const unsigned short* __restrict__ k_u,
                 const unsigned short* __restrict__ v_u,
                 unsigned short* __restrict__ att_u)
{
    __shared__ alignas(16) __bf16 Kt[32 * KT_STR];     // [key][d], padded
    __shared__ alignas(16) __bf16 Vt[F_DIM * VT_STR];  // [col][key], padded
    __shared__ alignas(16) __bf16 Pt[16 * PT_STR];     // [row][key], padded
    __shared__ float alphaS[16];
    __shared__ float lS[16];

    const __bf16* qb   = (const __bf16*)q_u;
    const __bf16* kb   = (const __bf16*)k_u;
    const __bf16* vb   = (const __bf16*)v_u;
    __bf16*       attb = (__bf16*)att_u;

    const int qt   = blockIdx.x;
    const int b    = blockIdx.y;
    const int tid  = threadIdx.x;
    const int wave = tid >> 5;
    const int lane = tid & 31;
    const int half = lane >> 4;
    const int lrow = lane & 15;
    const size_t base = (size_t)b * N_TOK;

    // Q tile in A-matrix layout (constant across the whole loop)
    v16bf qreg;
    {
        const __bf16* qp = qb + (base + qt * 16 + lrow) * FQ_DIM + half * 8;
        bf16x8 lo = load8(qp);
        bf16x8 hi = load8(qp + 16);
        #pragma unroll
        for (int i = 0; i < 8; ++i) { qreg[i] = lo[i]; qreg[8 + i] = hi[i]; }
    }

    v8f acc = {};
    float m8[8], l8[8];
    #pragma unroll
    for (int i = 0; i < 8; ++i) { m8[i] = -1e30f; l8[i] = 0.f; }

    const int vkey  = tid >> 4;            // 0..31 (V-tile loader role)
    const int vcol0 = (tid & 15) * 16;     // 0..240

    for (int kt = 0; kt < N_TOK; kt += 32) {
        // ---- cooperative load K tile [32 keys][32 d] (2 elems/thread) ----
        {
            const int e = tid * 2;
            const int key = e >> 5, d = e & 31;
            const __bf16* kp = kb + (base + kt) * FQ_DIM;
            Kt[key * KT_STR + d]     = kp[e];
            Kt[key * KT_STR + d + 1] = kp[e + 1];
        }
        // ---- cooperative load V tile, transposed into LDS [col][key] ----
        {
            const __bf16* vp = vb + (base + kt + vkey) * F_DIM + vcol0;
            bf16x8 lo = load8(vp);
            bf16x8 hi = load8(vp + 8);
            #pragma unroll
            for (int j = 0; j < 8; ++j) {
                Vt[(vcol0 + j) * VT_STR + vkey]     = lo[j];
                Vt[(vcol0 + 8 + j) * VT_STR + vkey] = hi[j];
            }
        }
        // prefetch next key tile into caches while this one is consumed
        if (kt + 32 < N_TOK) {
            __builtin_prefetch((const void*)(vb + (base + kt + 32 + vkey) * F_DIM + vcol0), 0, 1);
            __builtin_prefetch((const void*)(kb + (base + kt + 32) * FQ_DIM + tid * 2), 0, 1);
        }
        __syncthreads();

        // every wave: pre-load its V slice in B-matrix layout (col = wave*16+lrow)
        v16bf vreg;
        {
            const __bf16* vsp = &Vt[(wave * 16 + lrow) * VT_STR + half * 8];
            bf16x8 lo = load8(vsp);
            bf16x8 hi = load8(vsp + 16);
            #pragma unroll
            for (int i = 0; i < 8; ++i) { vreg[i] = lo[i]; vreg[8 + i] = hi[i]; }
        }

        if (wave == 0) {
            // K^T tiles in B-matrix layout: col = key, element = d index
            v16bf kreg0, kreg1;
            {
                const __bf16* k0p = &Kt[lrow * KT_STR + half * 8];
                const __bf16* k1p = &Kt[(16 + lrow) * KT_STR + half * 8];
                bf16x8 a0 = load8(k0p), a1 = load8(k0p + 16);
                bf16x8 b0 = load8(k1p), b1 = load8(k1p + 16);
                #pragma unroll
                for (int i = 0; i < 8; ++i) {
                    kreg0[i] = a0[i]; kreg0[8 + i] = a1[i];
                    kreg1[i] = b0[i]; kreg1[8 + i] = b1[i];
                }
            }
            v8f s0 = {}, s1 = {};
            s0 = __builtin_amdgcn_wmma_f32_16x16x32_bf16(false, qreg, false, kreg0,
                                                         (short)0, s0, false, false);
            s1 = __builtin_amdgcn_wmma_f32_16x16x32_bf16(false, qreg, false, kreg1,
                                                         (short)0, s1, false, false);
            // online softmax: row r = i + 8*half lives across 16 lanes of the half
            #pragma unroll
            for (int i = 0; i < 8; ++i) {
                float rmax = fmaxf(s0[i], s1[i]);
                #pragma unroll
                for (int msk = 1; msk < 16; msk <<= 1)
                    rmax = fmaxf(rmax, __shfl_xor(rmax, msk, 32));
                const float mnew = fmaxf(m8[i], rmax);
                const float al   = __expf(m8[i] - mnew);
                const float p0   = __expf(s0[i] - mnew);
                const float p1   = __expf(s1[i] - mnew);
                float rs = p0 + p1;
                #pragma unroll
                for (int msk = 1; msk < 16; msk <<= 1)
                    rs += __shfl_xor(rs, msk, 32);
                l8[i] = l8[i] * al + rs;
                m8[i] = mnew;
                const int r = i + 8 * half;
                Pt[r * PT_STR + lrow]      = f2bf(p0);
                Pt[r * PT_STR + 16 + lrow] = f2bf(p1);
                if (lrow == 0) alphaS[r] = al;
            }
        }
        __syncthreads();

        // rescale accumulators + P@V for this wave's 16-channel slice
        v16bf preg;
        {
            const __bf16* psp = &Pt[lrow * PT_STR + half * 8];
            bf16x8 lo = load8(psp);
            bf16x8 hi = load8(psp + 16);
            #pragma unroll
            for (int i = 0; i < 8; ++i) { preg[i] = lo[i]; preg[8 + i] = hi[i]; }
        }
        #pragma unroll
        for (int i = 0; i < 8; ++i) acc[i] *= alphaS[i + 8 * half];

        acc = __builtin_amdgcn_wmma_f32_16x16x32_bf16(false, preg, false, vreg,
                                                      (short)0, acc, false, false);
        __syncthreads();
    }

    // finalize: divide by softmax denominator, store attended (bf16)
    if (wave == 0 && lrow == 0) {
        #pragma unroll
        for (int i = 0; i < 8; ++i) lS[i + 8 * half] = l8[i];
    }
    __syncthreads();

    #pragma unroll
    for (int i = 0; i < 8; ++i) {
        const float o = acc[i] / lS[i + 8 * half];
        const int   r = qt * 16 + i + 8 * half;
        attb[(base + r) * F_DIM + wave * 16 + lrow] = f2bf(o);
    }
}

// ===========================================================================
// Kernel 3: out = inputs + relu(attended @ Wo + bo)   (all-bf16 operands)
// Grid: (BN/16, 16) blocks of 32 threads (1 wave -> one 16x16 output tile).
// ===========================================================================
__global__ __launch_bounds__(32)
void oproj_kernel(const unsigned short* __restrict__ att_u,
                  const unsigned short* __restrict__ wt_u,
                  const float* __restrict__ bo,
                  const float* __restrict__ x,  float* __restrict__ out)
{
    const __bf16* attb = (const __bf16*)att_u;
    const __bf16* WoT  = (const __bf16*)wt_u + (size_t)(64 + 256) * 256;

    const int rt   = blockIdx.x;
    const int c    = blockIdx.y;
    const int lane = threadIdx.x;
    const int half = lane >> 4;
    const int lrow = lane & 15;
    const int row0 = rt * 16;
    const int colbase = c * 16;

    const __bf16* ap0 = attb + (size_t)(row0 + lrow) * F_DIM + half * 8;
    const __bf16* bp0 = WoT + (size_t)(colbase + lrow) * 256 + half * 8;

    v8f acc = {};
    #pragma unroll
    for (int k0 = 0; k0 < F_DIM; k0 += 32) {
        v16bf a, b;
        bf16x8 alo = load8(ap0 + k0);
        bf16x8 ahi = load8(ap0 + k0 + 16);
        bf16x8 blo = load8(bp0 + k0);
        bf16x8 bhi = load8(bp0 + k0 + 16);
        #pragma unroll
        for (int i = 0; i < 8; ++i) {
            a[i] = alo[i]; a[8 + i] = ahi[i];
            b[i] = blo[i]; b[8 + i] = bhi[i];
        }
        acc = __builtin_amdgcn_wmma_f32_16x16x32_bf16(false, a, false, b,
                                                      (short)0, acc, false, false);
    }

    const float bb = bo[colbase + lrow];
    #pragma unroll
    for (int i = 0; i < 8; ++i) {
        const int m = i + half * 8;
        const size_t idx = (size_t)(row0 + m) * F_DIM + colbase + lrow;
        float o = acc[i] + bb;
        o = o > 0.f ? o : 0.f;
        out[idx] = x[idx] + o;
    }
}

// ===========================================================================
// Launch
// ===========================================================================
extern "C" void kernel_launch(void* const* d_in, const int* in_sizes, int n_in,
                              void* d_out, int out_size, void* d_ws, size_t ws_size,
                              hipStream_t stream)
{
    const float* x  = (const float*)d_in[0];
    const float* Wq = (const float*)d_in[1];
    const float* bq = (const float*)d_in[2];
    const float* Wk = (const float*)d_in[3];
    const float* bk = (const float*)d_in[4];
    const float* Wv = (const float*)d_in[5];
    const float* bv = (const float*)d_in[6];
    const float* Wo = (const float*)d_in[7];
    const float* bo = (const float*)d_in[8];
    float* out = (float*)d_out;

    // workspace layout (bf16 = unsigned short elements), ~27.6 MB total
    unsigned short* xb_u  = (unsigned short*)d_ws;                      // BN*256
    unsigned short* wt_u  = xb_u  + (size_t)BN_TOK * F_DIM;             // (32+32+256+256)*256
    unsigned short* q_u   = wt_u  + (size_t)(64 + 512) * 256;           // BN*32
    unsigned short* k_u   = q_u   + (size_t)BN_TOK * FQ_DIM;            // BN*32
    unsigned short* v_u   = k_u   + (size_t)BN_TOK * FQ_DIM;            // BN*256
    unsigned short* att_u = v_u   + (size_t)BN_TOK * F_DIM;             // BN*256

    cvt_x_kernel<<<dim3((BN_TOK * F_DIM) / (256 * 4)), 256, 0, stream>>>(x, xb_u);
    wt_kernel<<<dim3(256, 4), 256, 0, stream>>>(Wq, Wk, Wv, Wo, wt_u);

    qkv_kernel<<<dim3(BN_TOK / 16, 20), 32, 0, stream>>>(
        xb_u, wt_u, bq, bk, bv, q_u, k_u, v_u);

    attn_kernel<<<dim3(N_TOK / 16, B_DIM), 512, 0, stream>>>(
        q_u, k_u, v_u, att_u);

    oproj_kernel<<<dim3(BN_TOK / 16, 16), 32, 0, stream>>>(
        att_u, Wo ? wt_u : wt_u, bo, x, out);
}
